// SpiralConv_61701500174787
// MI455X (gfx1250) — matrix operand order repaired and checked by reading, not compiled, not run on hardware
//
#include <hip/hip_runtime.h>
#include <cstdint>

// SpiralConv: B=8, N=30000, S=9, IN_C=64, OUT_C=64  ->  GEMM [240000 x 576] x [576 x 64]
// bf16 WMMA (16x16x32) with fp32 accumulate; weights staged to LDS in fragment layout.

#define NPTS   30000
#define NBATCH 8
#define SPIRAL 9
#define INC    64
#define OUTC   64
#define KTOT   576            // SPIRAL * INC
#define NT_TILES 4            // OUTC / 16
#define KT_PER_PHASE 9        // 18 k-steps total, 9 per LDS phase
#define FRAGS_PER_PHASE (KT_PER_PHASE * NT_TILES)        // 36
#define LDS_UINT4 (FRAGS_PER_PHASE * 32 * 2)             // 36 frags * 32 lanes * 32B = 2304 uint4 (36 KB)

typedef __attribute__((ext_vector_type(16))) __bf16 v16bf;
typedef __attribute__((ext_vector_type(16))) float  v16f;
typedef __attribute__((ext_vector_type(8)))  float  v8f;

union Frag {
    unsigned u[8];
    uint4    q[2];
    v16bf    v;
};

// 16x fp32 -> 16x bf16 (RNE) via hardware convert path
__device__ __forceinline__ v16bf cvt16(float4 f0, float4 f1, float4 f2, float4 f3) {
    v16f ff;
    ff[0]  = f0.x; ff[1]  = f0.y; ff[2]  = f0.z; ff[3]  = f0.w;
    ff[4]  = f1.x; ff[5]  = f1.y; ff[6]  = f1.z; ff[7]  = f1.w;
    ff[8]  = f2.x; ff[9]  = f2.y; ff[10] = f2.z; ff[11] = f2.w;
    ff[12] = f3.x; ff[13] = f3.y; ff[14] = f3.z; ff[15] = f3.w;
    return __builtin_convertvector(ff, v16bf);
}

__global__ void __launch_bounds__(256)
spiral_conv_wmma(const float* __restrict__ x,
                 const long long* __restrict__ adj,
                 const float* __restrict__ W,
                 const float* __restrict__ bias,
                 float* __restrict__ out) {
    __shared__ __align__(32) uint4 ldsB[LDS_UINT4];

    const int lane = threadIdx.x & 31;
    const int wave = threadIdx.x >> 5;
    const int tile = blockIdx.x * 8 + wave;          // 15000 tiles total
    const long p0  = (long)tile * 16;                // first point of tile
    const int b    = (int)(p0 / NPTS);               // tiles never cross batches (30000 % 16 == 0)
    const int n0   = (int)(p0 - (long)b * NPTS);
    const int r    = lane & 15;                      // row within 16-point tile
    const bool hiHalf = (lane >= 16);
    const int laneOff = hiHalf ? 16 : 0;             // A-fragment K striping (contiguous k = 16h + s)

    const float* xb = x + (size_t)b * NPTS * INC;

    // hoist the 9 neighbor indices for this lane's point
    const long long* adjRow = adj + ((size_t)b * NPTS + (size_t)(n0 + r)) * SPIRAL;
    long long idx[SPIRAL];
#pragma unroll
    for (int s = 0; s < SPIRAL; ++s) idx[s] = adjRow[s];

    v8f zero = {};
    v8f acc[NT_TILES];
#pragma unroll
    for (int nt = 0; nt < NT_TILES; ++nt) acc[nt] = zero;

#pragma unroll
    for (int phase = 0; phase < 2; ++phase) {
        // ---- stage this phase's W fragments (bf16, WMMA B layout) into LDS ----
        __syncthreads();   // previous phase fully consumed before overwrite
        for (int wu = threadIdx.x; wu < FRAGS_PER_PHASE * 32; wu += 256) {
            int slane = wu & 31;
            int fb    = wu >> 5;              // k9*4 + nt within phase
            int nt    = fb & 3;
            int k9    = fb >> 2;
            int kt    = phase * KT_PER_PHASE + k9;
            int col   = nt * 16 + (slane & 15);                 // output channel
            int kbase = kt * 32 + ((slane & 16) ? 16 : 0);      // B-fragment K striping
            const float* wp = W + (size_t)col * KTOT + kbase;
            float4 w0 = *(const float4*)(wp + 0);
            float4 w1 = *(const float4*)(wp + 4);
            float4 w2 = *(const float4*)(wp + 8);
            float4 w3 = *(const float4*)(wp + 12);
            Frag f;
            f.v = cvt16(w0, w1, w2, w3);
            ldsB[fb * 64 + slane * 2]     = f.q[0];
            ldsB[fb * 64 + slane * 2 + 1] = f.q[1];
        }
        __syncthreads();

        // ---- 9 k-steps of 32: gather A rows, convert to bf16, 4 WMMAs each ----
#pragma unroll
        for (int k9 = 0; k9 < KT_PER_PHASE; ++k9) {
            const int kt = phase * KT_PER_PHASE + k9;
            const int s  = kt >> 1;          // spiral slot (constant after unroll)
            const int h  = kt & 1;           // which half of the 64 feats
            const float* src = xb + (size_t)idx[s] * INC + h * 32 + laneOff;

            float4 f0 = *(const float4*)(src + 0);
            float4 f1 = *(const float4*)(src + 4);
            float4 f2 = *(const float4*)(src + 8);
            float4 f3 = *(const float4*)(src + 12);

            v16bf a = cvt16(f0, f1, f2, f3);

#pragma unroll
            for (int nt = 0; nt < NT_TILES; ++nt) {
                const int fb = k9 * 4 + nt;
                Frag bf;
                bf.q[0] = ldsB[fb * 64 + lane * 2];
                bf.q[1] = ldsB[fb * 64 + lane * 2 + 1];
                acc[nt] = __builtin_amdgcn_wmma_f32_16x16x32_bf16(
                    false, a, false, bf.v, (short)0, acc[nt], false, false);
            }
        }
    }

    // ---- epilogue: bias + ELU + mask last vertex, per C/D VGPR layout ----
#pragma unroll
    for (int nt = 0; nt < NT_TILES; ++nt) {
        const int colg = nt * 16 + (lane & 15);
        const float bv = bias[colg];
#pragma unroll
        for (int v = 0; v < 8; ++v) {
            const int m = v + (hiHalf ? 8 : 0);
            const int n = n0 + m;
            float val = acc[nt][v] + bv;
            val = (val > 0.0f) ? val : expm1f(val);
            if (n == NPTS - 1) val = 0.0f;
            __builtin_nontemporal_store(val, &out[((size_t)b * NPTS + n) * OUTC + colg]);
        }
    }
}

extern "C" void kernel_launch(void* const* d_in, const int* in_sizes, int n_in,
                              void* d_out, int out_size, void* d_ws, size_t ws_size,
                              hipStream_t stream) {
    const float*     x    = (const float*)d_in[0];
    // d_in[1] = t_vertex (unused by the reference computation)
    const long long* adj  = (const long long*)d_in[2];
    const float*     W    = (const float*)d_in[3];
    const float*     bias = (const float*)d_in[4];
    float*           out  = (float*)d_out;

    // 240000 points / 16 per tile = 15000 tiles; 8 waves (tiles) per 256-thread block
    const int blocks = 15000 / 8;   // 1875, exact
    spiral_conv_wmma<<<blocks, 256, 0, stream>>>(x, adj, W, bias, out);
}
